// MyRNN_79096117723343
// MI455X (gfx1250) — compile-verified, matrix operand order checked by reading, not run on hardware
//
#include <hip/hip_runtime.h>
#include <hip/hip_bf16.h>
#include <math.h>

// ---------------------------------------------------------------------------
// MyRNN on MI455X (gfx1250, wave32, WMMA, TDM)
//   Phase A: wx = x @ W_w^T + W_b   (32768x1024x1024 GEMM, bf16 WMMA, f32 acc)
//   Phase B: persistent flag-synced scan: t = U_w @ tanh(wx_t + t) + U_b
//            d_out[:, t, :] doubles as the cross-step state buffer; the wx
//            tile for step t+1 is prefetched into LDS by the Tensor Data
//            Mover (tensor_load_to_lds) while step t's WMMAs run.
// ---------------------------------------------------------------------------

typedef __bf16 bf16;
typedef __attribute__((ext_vector_type(16))) __bf16        v16bf;
typedef __attribute__((ext_vector_type(8)))  __bf16        v8bf;
typedef __attribute__((ext_vector_type(8)))  float         v8f;
typedef __attribute__((ext_vector_type(4)))  unsigned int  v4u;
typedef __attribute__((ext_vector_type(8)))  unsigned int  v8u;

#define B_   64
#define S_   512
#define E_   1024
#define H_   1024
#define BS_  (B_ * S_)                 // 32768
static const size_t TFINAL_OFF = (size_t)B_ * S_ * H_;   // 33,554,432 floats

// ---- fast tanh: gfx1250 has V_TANH_F32 (transcendental VALU, co-executes
// with WMMA). Guarded so a name mismatch degrades to libm, never breaks. ----
#if __has_builtin(__builtin_amdgcn_tanhf)
__device__ __forceinline__ float fast_tanh(float x) { return __builtin_amdgcn_tanhf(x); }
#elif __has_builtin(__builtin_amdgcn_tanh_f32)
__device__ __forceinline__ float fast_tanh(float x) { return __builtin_amdgcn_tanh_f32(x); }
#else
__device__ __forceinline__ float fast_tanh(float x) { return tanhf(x); }
#endif

// ---- WMMA helpers ---------------------------------------------------------

__device__ __forceinline__ v8f wmma_bf16(v16bf a, v16bf b, v8f c) {
  // v_wmma_f32_16x16x32_bf16  D = A(16x32) * B(32x16) + C(16x16 f32)
  return __builtin_amdgcn_wmma_f32_16x16x32_bf16(
      /*neg_a=*/false, a, /*neg_b=*/false, b,
      /*c_mod=*/(short)0, c, /*reuse_a=*/false, /*reuse_b=*/false);
}

// Load one 16x32 bf16 operand fragment from an LDS tile stored row-major
// with row stride `rs` (elements). Per ISA layout:
//   lanes 0-15 : row = lane,    K in {0..7} u {16..23}
//   lanes 16-31: row = lane-16, K in {8..15} u {24..31}
// i.e. two contiguous 16-byte chunks per lane -> two ds_read_b128.
__device__ __forceinline__ v16bf load_frag(const bf16* tile, int rs) {
  const int lane = threadIdx.x & 31;
  const int r    = lane & 15;
  const int h    = lane >> 4;
  const bf16* p  = tile + r * rs + h * 8;
  v8bf lo = *(const v8bf*)(p);
  v8bf hi = *(const v8bf*)(p + 16);
  return __builtin_shufflevector(lo, hi, 0,1,2,3,4,5,6,7,8,9,10,11,12,13,14,15);
}

// ---- Tensor Data Mover: async 2-D bf16 tile load, global -> LDS -----------
// Builds a 2-group D# (ISA cdna5 §8.3/8.4) and issues TENSOR_LOAD_TO_LDS via
// inline asm (portable across the 5-arg/6-arg builtin arity split).
//   lds_off          : byte offset of destination within the WG's LDS
//   gptr             : global address of tile start
//   width_elems      : tile_dim0 == tensor_dim0 (bf16 elements per row)
//   rows             : tile_dim1
//   row_stride_elems : tensor_dim0_stride (bf16 elements between rows)
__device__ __forceinline__ void tdm_load_2d_bf16(unsigned lds_off,
                                                 const void* gptr,
                                                 unsigned width_elems,
                                                 unsigned rows,
                                                 unsigned long long row_stride_elems) {
  const unsigned long long ga = (unsigned long long)gptr;
  v4u g0;
  g0[0] = 1u;                                            // count=1, is_restore=0
  g0[1] = lds_off;                                       // lds_addr (bytes)
  g0[2] = (unsigned)(ga & 0xFFFFFFFFu);                  // global_addr[31:0]
  g0[3] = (unsigned)((ga >> 32) & 0x1FFFFFFu) | (2u << 30);  // [56:32] | type=2

  const unsigned td0 = width_elems;        // tensor_dim0 (row length bound)
  const unsigned td1 = 0x40000000u;        // tensor_dim1: large (no row OOB)
  v8u g1;
  g1[0] = (1u << 16);                                    // data_size=1 (2 bytes)
  g1[1] = (td0 & 0xFFFFu) << 16;                         // tensor_dim0[15:0]
  g1[2] = (td0 >> 16) | ((td1 & 0xFFFFu) << 16);         // td0 hi | td1 lo
  g1[3] = (td1 >> 16) | ((width_elems & 0xFFFFu) << 16); // td1 hi | tile_dim0
  g1[4] = rows & 0xFFFFu;                                // tile_dim1; tile_dim2=0
  g1[5] = (unsigned)(row_stride_elems & 0xFFFFFFFFull);  // dim0_stride[31:0]
  g1[6] = (unsigned)((row_stride_elems >> 32) & 0xFFFFull); // dim0_stride[47:32]
  g1[7] = 0u;                                            // dim1_stride unused
  asm volatile("tensor_load_to_lds %0, %1" :: "s"(g0), "s"(g1) : "memory");
}

// ---------------------------------------------------------------------------
// Kernel Z: zero the step-completion flags (stream-ordered prologue so every
// kernel_launch call is deterministic; harness does not re-poison d_ws).
// ---------------------------------------------------------------------------
__global__ void zero_flags_kernel(int* __restrict__ flags, int n) {
  int i = blockIdx.x * blockDim.x + threadIdx.x;
  if (i < n) flags[i] = 0;
}

// ---------------------------------------------------------------------------
// Kernel A: wx[m, n] = sum_k x[m,k] * W_w[n,k] + W_b[n],   stored as bf16.
// Block = 256 threads (8 waves). Block tile 128(M) x 128(N), K-step 32.
// Each wave: 4 M-tiles x 2 N-tiles = 8 f32 accumulators.
// ---------------------------------------------------------------------------
#define TM 128
#define TN 128
#define TK 32
#define KP 40          // padded LDS row stride (elements), keeps 16B alignment

__global__ __launch_bounds__(256) void wx_gemm_kernel(
    const float* __restrict__ x,    // [BS_, E_]
    const float* __restrict__ Ww,   // [H_, E_]
    const float* __restrict__ Wb,   // [H_]
    bf16* __restrict__ wx)          // [BS_, H_]
{
  __shared__ bf16 As[TM * KP];
  __shared__ bf16 Bs[TN * KP];

  const int m0   = blockIdx.y * TM;
  const int n0   = blockIdx.x * TN;
  const int tid  = threadIdx.x;
  const int wave = tid >> 5;
  const int lane = tid & 31;
  const int wm   = (wave >> 2) * 64;   // wave M offset: 0 or 64
  const int wn   = (wave & 3) * 32;    // wave N offset: 0,32,64,96

  v8f acc[4][2] = {};

  for (int k0 = 0; k0 < E_; k0 += TK) {
    if (k0 + TK < E_) {   // speculative prefetch of next K-slab (global_prefetch)
      __builtin_prefetch(x  + (size_t)(m0 + (tid >> 1)) * E_ + k0 + TK, 0, 1);
      __builtin_prefetch(Ww + (size_t)(n0 + (tid >> 1)) * E_ + k0 + TK, 0, 1);
    }
    // Stage A (x) and B (W_w) tiles into LDS, f32 -> bf16 on the fly.
    // 128 rows x 8 float4 each = 1024 float4 per tile; 4 per thread.
#pragma unroll
    for (int j = 0; j < 4; ++j) {
      const int idx = tid * 4 + j;        // 0..1023
      const int row = idx >> 3;           // 0..127
      const int c4  = idx & 7;            // 0..7
      const float4 va = *(const float4*)(x  + (size_t)(m0 + row) * E_ + k0 + c4 * 4);
      bf16* da = &As[row * KP + c4 * 4];
      da[0] = (bf16)va.x; da[1] = (bf16)va.y; da[2] = (bf16)va.z; da[3] = (bf16)va.w;
      const float4 vb = *(const float4*)(Ww + (size_t)(n0 + row) * E_ + k0 + c4 * 4);
      bf16* db = &Bs[row * KP + c4 * 4];
      db[0] = (bf16)vb.x; db[1] = (bf16)vb.y; db[2] = (bf16)vb.z; db[3] = (bf16)vb.w;
    }
    __syncthreads();

    v16bf afr[4], bfr[2];
#pragma unroll
    for (int i = 0; i < 4; ++i) afr[i] = load_frag(&As[(wm + i * 16) * KP], KP);
#pragma unroll
    for (int j = 0; j < 2; ++j) bfr[j] = load_frag(&Bs[(wn + j * 16) * KP], KP);
#pragma unroll
    for (int i = 0; i < 4; ++i)
#pragma unroll
      for (int j = 0; j < 2; ++j)
        acc[i][j] = wmma_bf16(afr[i], bfr[j], acc[i][j]);

    __syncthreads();
  }

  // Epilogue: += W_b[n], convert to bf16, store.
  // C layout: lanes 0-15 -> N=lane, M=i; lanes 16-31 -> N=lane-16, M=8+i.
  const int half = lane >> 4;
  const int nl   = lane & 15;
#pragma unroll
  for (int j = 0; j < 2; ++j) {
    const int n     = n0 + wn + j * 16 + nl;
    const float bia = Wb[n];
#pragma unroll
    for (int i = 0; i < 4; ++i) {
      const int mrow = m0 + wm + i * 16 + half * 8;
#pragma unroll
      for (int e = 0; e < 8; ++e)
        wx[(size_t)(mrow + e) * H_ + n] = (bf16)(acc[i][j][e] + bia);
    }
  }
}

// ---------------------------------------------------------------------------
// Kernel B: persistent scan. 64 WGs = 4 batch-tiles (M=16) x 16 N-slabs (64).
// Each WG keeps its 64x1024 bf16 slab of U_w in LDS (129 KB); per step the
// TDM has already deposited wx[btile, t, :] (16x1024 bf16, unpadded rows) in
// a double-buffered LDS region. The WG builds h3 = tanh(wx_t + t_prev)
// (single v_tanh_f32 per element, co-executes with WMMA), runs the WMMA
// K-loop, publishes its 16x64 f32 tile into d_out, and the TDM load for
// step t+2 is issued as soon as the h3 build retires.
// Cross-WG step ordering: release atomicAdd on flags[btile][t], acquire spin.
// ---------------------------------------------------------------------------
#define HP 1032   // padded row stride for 1024-wide LDS rows (U slab, h3)

// LDS byte-offset map (dynamic LDS, offset 0 == smem base)
#define ULDS_OFF   0u
#define H3_OFF     (64u * HP * 2u)               // 132,096
#define WXBUF_OFF  (H3_OFF + 16u * HP * 2u)      // 165,120
#define WXBUF_SZ   (16u * 1024u * 2u)            // 32,768 per buffer
#define LDS_TOTAL  (WXBUF_OFF + 2u * WXBUF_SZ)   // 230,656 B (< 320KB/WGP)

__global__ __launch_bounds__(128) void rnn_scan_kernel(
    const bf16*  __restrict__ wx,    // [BS_, H_]
    const float* __restrict__ Uw,    // [H_, H_]
    const float* __restrict__ Ub,    // [H_]
    float*       __restrict__ out,   // result1 [B,S,H] then t_final [B,H]
    int*         __restrict__ flags) // [4 * S_]
{
  extern __shared__ bf16 smem[];
  bf16* Ulds  = smem;                         // [64][HP]
  bf16* h3    = smem + 64 * HP;               // [16][HP]
  bf16* wxbuf = smem + 64 * HP + 16 * HP;     // [2][16][1024] (TDM-filled)

  const int g     = blockIdx.x;
  const int btile = g >> 4;           // 0..3
  const int nslab = g & 15;           // 0..15
  const int n0    = nslab * 64;
  const int tid   = threadIdx.x;
  const int wave  = tid >> 5;
  const int lane  = tid & 31;
  const int b0    = btile * 16;       // first batch row of this tile

  // ---- kick off TDM prefetch of wx tiles for steps 0 and 1 ----
  // 16 rows (batch), each H_ bf16 wide, rows strided by S_*H_ elements.
  if (wave == 0) {
    tdm_load_2d_bf16(WXBUF_OFF,            wx + (size_t)b0 * S_ * H_ + 0 * (size_t)H_,
                     H_, 16, (unsigned long long)S_ * H_);
    tdm_load_2d_bf16(WXBUF_OFF + WXBUF_SZ, wx + (size_t)b0 * S_ * H_ + 1 * (size_t)H_,
                     H_, 16, (unsigned long long)S_ * H_);
  }

  // ---- preload U slab (rows n0..n0+63, all K) into LDS as bf16 ----
  for (int i = tid; i < 64 * 256; i += 128) {   // 64 rows x 256 float4
    const int row = i >> 8;
    const int c4  = i & 255;
    const float4 v = *(const float4*)(Uw + (size_t)(n0 + row) * H_ + c4 * 4);
    bf16* d = &Ulds[row * HP + c4 * 4];
    d[0] = (bf16)v.x; d[1] = (bf16)v.y; d[2] = (bf16)v.z; d[3] = (bf16)v.w;
  }
  __syncthreads();

  const int   nw    = n0 + wave * 16;           // this wave's N tile base
  const int   nl    = lane & 15;
  const int   half  = lane >> 4;
  const float ubias = Ub[nw + nl];

  for (int t = 0; t < S_; ++t) {
    // ---- wait for step t-1 of this batch tile to be fully published ----
    if (t > 0) {
      if (tid == 0) {
        while (__hip_atomic_load(&flags[btile * S_ + t - 1],
                                 __ATOMIC_ACQUIRE, __HIP_MEMORY_SCOPE_AGENT) < 16) {
          __builtin_amdgcn_s_sleep(1);
        }
      }
      __syncthreads();
    }

    // ---- wave 0 owns TENSORcnt: ensure the TDM load for step t landed ----
    if (wave == 0) {
      if (t + 1 < S_) __builtin_amdgcn_s_wait_tensorcnt(1);  // load t done
      else            __builtin_amdgcn_s_wait_tensorcnt(0);  // last step
    }
    __syncthreads();

    // ---- h3 = tanh(wxbuf[t&1] + t_prev) -> LDS (bf16) ----
    // wxbuf rows have unpadded 2048B stride (TDM fill order), so use
    // lane-contiguous k = it*256 + lane*8 addressing (bank-conflict-free).
    {
      const bf16* wb = wxbuf + (t & 1) * (16 * 1024);
      if (t > 0) {
#pragma unroll
        for (int rr = 0; rr < 4; ++rr) {
          const int r = wave * 4 + rr;
          const float* tp = out + ((size_t)(b0 + r) * S_ + (t - 1)) * H_;
#pragma unroll 2
          for (int it = 0; it < 4; ++it) {
            const int k = it * 256 + lane * 8;
            v8bf wv = *(const v8bf*)(wb + r * 1024 + k);
            v8f  tv = *(const v8f*)(tp + k);
            v8bf hv;
#pragma unroll
            for (int e = 0; e < 8; ++e) hv[e] = (bf16)fast_tanh((float)wv[e] + tv[e]);
            *(v8bf*)(&h3[r * HP + k]) = hv;
          }
        }
      } else {
#pragma unroll
        for (int rr = 0; rr < 4; ++rr) {
          const int r = wave * 4 + rr;
#pragma unroll 2
          for (int it = 0; it < 4; ++it) {
            const int k = it * 256 + lane * 8;
            v8bf wv = *(const v8bf*)(wb + r * 1024 + k);
            v8bf hv;
#pragma unroll
            for (int e = 0; e < 8; ++e) hv[e] = (bf16)fast_tanh((float)wv[e]);
            *(v8bf*)(&h3[r * HP + k]) = hv;
          }
        }
      }
    }
    __syncthreads();   // all waves done reading wxbuf[t&1] + h3 complete

    // ---- buffer free: issue TDM load for step t+2 into wxbuf[t&1] ----
    if (wave == 0 && t + 2 < S_) {
      tdm_load_2d_bf16(WXBUF_OFF + (unsigned)(t & 1) * WXBUF_SZ,
                       wx + (size_t)b0 * S_ * H_ + (size_t)(t + 2) * H_,
                       H_, 16, (unsigned long long)S_ * H_);
    }

    // ---- 16x16 output tile per wave: K = 1024 in 32 chunks of 32.
    //      Two interleaved accumulators break the WMMA RAW chain (ILP=2). ----
    v8f acc0 = {}, acc1 = {};
#pragma unroll 4
    for (int kk = 0; kk < 32; kk += 2) {
      v16bf a0 = load_frag(&h3[kk * 32], HP);
      v16bf bq0 = load_frag(&Ulds[(wave * 16) * HP + kk * 32], HP);
      acc0 = wmma_bf16(a0, bq0, acc0);
      v16bf a1 = load_frag(&h3[(kk + 1) * 32], HP);
      v16bf bq1 = load_frag(&Ulds[(wave * 16) * HP + (kk + 1) * 32], HP);
      acc1 = wmma_bf16(a1, bq1, acc1);
    }
    v8f c = acc0 + acc1 + ubias;

    // ---- publish t_new tile into d_out (this IS the reference output) ----
    const int n = nw + nl;
#pragma unroll
    for (int e = 0; e < 8; ++e) {
      const int m = half * 8 + e;
      const int b = b0 + m;
      const float val = c[e];
      out[((size_t)b * S_ + t) * H_ + n] = val;
      if (t == S_ - 1) out[TFINAL_OFF + (size_t)b * H_ + n] = val;  // t_final
    }
    __threadfence();        // device-scope release of this WG's stores
    __syncthreads();
    if (tid == 0)
      __hip_atomic_fetch_add(&flags[btile * S_ + t], 1,
                             __ATOMIC_RELEASE, __HIP_MEMORY_SCOPE_AGENT);
  }
}

// ---------------------------------------------------------------------------
// kernel_launch
//   d_in: [emben_x (B,S,E) f32][W_w (H,E)][W_b (H)][U_w (H,H)][U_b (H)]
//   d_out: result1 (B,S,H) f32  ++  t_final (B,H) f32
//   d_ws: wx bf16 [BS_*H_] (64 MB) + flags [4*S_] ints
// ---------------------------------------------------------------------------
extern "C" void kernel_launch(void* const* d_in, const int* in_sizes, int n_in,
                              void* d_out, int out_size, void* d_ws, size_t ws_size,
                              hipStream_t stream) {
  const float* x  = (const float*)d_in[0];
  const float* Ww = (const float*)d_in[1];
  const float* Wb = (const float*)d_in[2];
  const float* Uw = (const float*)d_in[3];
  const float* Ub = (const float*)d_in[4];
  float* out = (float*)d_out;

  bf16* wx   = (bf16*)d_ws;
  int* flags = (int*)((char*)d_ws + (size_t)BS_ * H_ * sizeof(bf16));

  // Allow >64KB dynamic LDS for the scan kernel (gfx1250: 320KB/WGP).
  (void)hipFuncSetAttribute(reinterpret_cast<const void*>(rnn_scan_kernel),
                            hipFuncAttributeMaxDynamicSharedMemorySize,
                            (int)LDS_TOTAL);

  zero_flags_kernel<<<8, 256, 0, stream>>>(flags, 4 * S_);

  dim3 gA(H_ / TN, BS_ / TM);   // (8, 256)
  wx_gemm_kernel<<<gA, 256, 0, stream>>>(x, Ww, Wb, wx);

  rnn_scan_kernel<<<64, 128, LDS_TOTAL, stream>>>(wx, Uw, Ub, out, flags);
}
